// BiLSTM_CRF_9182640079410
// MI455X (gfx1250) — matrix operand order repaired and checked by reading, not compiled, not run on hardware
//
#include <hip/hip_runtime.h>
#include <hip/hip_bf16.h>
#include <math.h>

// ---------------------------------------------------------------------------
// BiLSTM-CRF for MI455X (gfx1250, wave32, WMMA).
// B=64, L=512, V=30000, E=256, H=256 (H2=128), T=9.
// ---------------------------------------------------------------------------

typedef __attribute__((ext_vector_type(16))) __bf16        v16bf;
typedef __attribute__((ext_vector_type(8)))  float         v8f;
typedef __attribute__((ext_vector_type(4)))  unsigned int  u32x4;

union Frag16 {            // 32 bytes: one bf16 WMMA A/B fragment (8 VGPRs)
    u32x4 u[2];
    v16bf v;
};
union Frag32 {            // 32 bytes: one f32 C/D fragment (8 VGPRs)
    u32x4 u[2];
    v8f   f;
};

__device__ __forceinline__ unsigned short f32_to_bf16(float f) {
    union { float f; unsigned u; } x; x.f = f;
    unsigned r = x.u + 0x7FFFu + ((x.u >> 16) & 1u);   // round-to-nearest-even
    return (unsigned short)(r >> 16);
}
__device__ __forceinline__ float bf16_to_f32(unsigned short h) {
    union { unsigned u; float f; } x; x.u = ((unsigned)h) << 16; return x.f;
}

// Problem constants
#define BSZ  64
#define LSEQ 512
#define EMB  256
#define H2   128
#define NG   1024              // 4*H2 * 2 directions (gate columns, fwd|bwd)
#define NT   9
#define ML   (BSZ * LSEQ)      // 32768 rows, time-major m = l*64 + b

// ---------------- workspace layout (bytes) ----------------
#define ALN(x) (((x) + 255) & ~(size_t)255)
static const size_t OFF_X    = 0;                                  // bf16 [ML][EMB]
static const size_t SZ_X     = (size_t)ML * EMB * 2;
static const size_t OFF_WIH  = ALN(OFF_X + SZ_X);                  // bf16 [NG][EMB]   (K-major, n: fwd|bwd)
static const size_t SZ_WIH   = (size_t)EMB * NG * 2;
static const size_t OFF_WHH  = ALN(OFF_WIH + SZ_WIH);              // bf16 [2][512][H2] (K-major per dir)
static const size_t SZ_WHH   = (size_t)2 * H2 * 512 * 2;
static const size_t OFF_BIAS = ALN(OFF_WHH + SZ_WHH);              // f32 [NG]
static const size_t SZ_BIAS  = (size_t)NG * 4;
// gx stored FRAGMENT-TILED: tileIdx = ((l*2 + dir)*32 + nT)*4 + mT,
// each tile = 256 f32 in lane-major fragment order (lane*8 + r).
static const size_t OFF_GX   = ALN(OFF_BIAS + SZ_BIAS);            // f32 [ML*NG] tiled
static const size_t SZ_GX    = (size_t)ML * NG * 4;
static const size_t OFF_LO   = ALN(OFF_GX + SZ_GX);                // f32 [ML][256] lstm_out (fwd|bwd)
static const size_t SZ_LO    = (size_t)ML * 256 * 4;
static const size_t OFF_EMI  = ALN(OFF_LO + SZ_LO);                // f32 [ML][NT]
static const size_t SZ_EMI   = (size_t)ML * NT * 4;
static const size_t OFF_HIST = ALN(OFF_EMI + SZ_EMI);              // u8 [BSZ][LSEQ-1][NT]

// ---------------------------------------------------------------------------
// Kernel 1: weight conversion / transposition to WMMA-friendly bf16 layouts.
// ---------------------------------------------------------------------------
__global__ void prep_kernel(const float* __restrict__ wihf, const float* __restrict__ whhf,
                            const float* __restrict__ bf,   const float* __restrict__ wihb,
                            const float* __restrict__ whhb, const float* __restrict__ bb,
                            unsigned short* __restrict__ WihT,
                            unsigned short* __restrict__ WhhT,
                            float* __restrict__ biasG) {
    int gid = blockIdx.x * blockDim.x + threadIdx.x;
    if (gid < EMB * NG) {                       // 262144
        int e = gid >> 10, n = gid & 1023;
        float v = (n < 512) ? wihf[n * EMB + e] : wihb[(n - 512) * EMB + e];
        WihT[n * EMB + e] = f32_to_bf16(v);
    } else if (gid < EMB * NG + 2 * H2 * 512) { // +131072
        int id  = gid - EMB * NG;
        int dir = id >> 16;
        int r   = id & 65535;
        int k   = r >> 9, n = r & 511;
        const float* w = dir ? whhb : whhf;
        WhhT[dir * 65536 + n * H2 + k] = f32_to_bf16(w[n * H2 + k]);
    } else if (gid < EMB * NG + 2 * H2 * 512 + NG) {
        int n = gid - (EMB * NG + 2 * H2 * 512);
        biasG[n] = (n < 512) ? bf[n] : bb[n - 512];
    }
}

// ---------------------------------------------------------------------------
// Kernel 2: embedding gather -> time-major bf16 X[l*64+b][e]
// ---------------------------------------------------------------------------
__global__ void embed_kernel(const int* __restrict__ sentence,
                             const float* __restrict__ emb,
                             unsigned short* __restrict__ X) {
    int m = blockIdx.x, e = threadIdx.x;
    int l = m >> 6, b = m & 63;
    int tok = sentence[b * LSEQ + l];
    X[(size_t)m * EMB + e] = f32_to_bf16(emb[(size_t)tok * EMB + e]);
}

// ---------------------------------------------------------------------------
// Kernel 3: input GEMM  gx = X @ WihT + bias.  M=32768, N=1024, K=256.
// Block = 256 thr (8 waves). Each wave owns a 32x64 strip: TWO m-tiles
// (A reuse of every B fragment) x 4 n-tiles. Output stored fragment-tiled
// (2x b128 per tile) so the LSTM kernel can load accumulators with b128s.
// ---------------------------------------------------------------------------
__launch_bounds__(256)
__global__ void gemm_gx_kernel(const unsigned short* __restrict__ X,
                               const unsigned short* __restrict__ WihT,
                               const float* __restrict__ biasG,
                               float* __restrict__ gx) {
    const int lane  = threadIdx.x & 31;
    const int wave  = threadIdx.x >> 5;
    const int mPair = blockIdx.x;                        // 0..1023 (two 16-row tiles)
    const int nBase = blockIdx.y * 512 + wave * 64;      // 0..960
    const int hiA   = (lane >> 4) << 3;                  // A K-offset: 0 or 8
    const int rowA0 = mPair * 32 + (lane & 15);
    const int colC  = lane & 15;

    v8f c[2][4];
#pragma unroll
    for (int s = 0; s < 4; ++s) {
        float bv = biasG[nBase + s * 16 + colC];
#pragma unroll
        for (int r = 0; r < 8; ++r) { c[0][s][r] = bv; c[1][s][r] = bv; }
    }
#pragma unroll 2
    for (int kk = 0; kk < 8; ++kk) {
        Frag16 a0, a1;
        const unsigned short* ap0 = X + (size_t)rowA0 * EMB + kk * 32 + hiA;
        a0.u[0] = *(const u32x4*)(ap0);
        a0.u[1] = *(const u32x4*)(ap0 + 16);
        const unsigned short* ap1 = ap0 + 16 * EMB;
        a1.u[0] = *(const u32x4*)(ap1);
        a1.u[1] = *(const u32x4*)(ap1 + 16);
        const int kB = kk * 32 + hiA;                    // B: lane = N column
#pragma unroll
        for (int s = 0; s < 4; ++s) {
            Frag16 bfr;
            const unsigned short* bp = WihT + (size_t)(nBase + s * 16 + colC) * EMB + kB;
            bfr.u[0] = *(const u32x4*)(bp);
            bfr.u[1] = *(const u32x4*)(bp + 16);
            c[0][s] = __builtin_amdgcn_wmma_f32_16x16x32_bf16(
                false, a0.v, false, bfr.v, (short)0, c[0][s], false, false);
            c[1][s] = __builtin_amdgcn_wmma_f32_16x16x32_bf16(
                false, a1.v, false, bfr.v, (short)0, c[1][s], false, false);
        }
    }
    // fragment-tiled store: tileIdx = ((l*2 + dir)*32 + nT)*4 + mT
#pragma unroll
    for (int w2 = 0; w2 < 2; ++w2) {
        const int mTg = mPair * 2 + w2;                  // global 16-row tile
        const int l   = mTg >> 2;
        const int mT  = mTg & 3;
#pragma unroll
        for (int s = 0; s < 4; ++s) {
            const int ncol = nBase + s * 16;
            const int dir  = ncol >> 9;
            const int nT   = (ncol & 511) >> 4;
            const size_t tileIdx = (((size_t)l * 2 + dir) * 32 + nT) * 4 + mT;
            float* p = gx + tileIdx * 256 + lane * 8;
            Frag32 cf; cf.f = c[w2][s];
            *(u32x4*)(p)     = cf.u[0];
            *(u32x4*)(p + 4) = cf.u[1];
        }
    }
}

// ---------------------------------------------------------------------------
// Kernel 4: sequential LSTM, one workgroup per direction (grid=2, 512 thr).
// LDS: h (bf16 16KB) | c (f32 32KB) | gates (f32 128KB) | WhhT (bf16 128KB)
// = 304KB of the 320KB WGP LDS.  w_hh staged via async global->LDS B128.
// Next step's gx slice (contiguous 128KB, fragment-tiled) is prefetched.
// ---------------------------------------------------------------------------
__launch_bounds__(512)
__global__ void lstm_kernel(const float* __restrict__ gx,
                            const unsigned short* __restrict__ WhhT,
                            const int* __restrict__ pm,
                            float* __restrict__ lstm_out) {
    extern __shared__ char smem[];
    unsigned short* hS = (unsigned short*)smem;                           // 16384 B
    float*          cS = (float*)(smem + 16384);                          // 32768 B
    float*          gS = (float*)(smem + 16384 + 32768);                  // 131072 B
    unsigned short* wS = (unsigned short*)(smem + 16384 + 32768 + 131072);// 131072 B

    const int dir  = blockIdx.x;
    const int tid  = threadIdx.x;
    const int lane = tid & 31;
    const int wave = tid >> 5;

    // ---- stage w_hh^T into LDS with CDNA5 async global->LDS copies ----
    {
        const char* src = (const char*)(WhhT + (size_t)dir * H2 * 512);
        const unsigned ldsBase = (unsigned)(unsigned long long)(void*)wS;
        for (int i = tid; i < (H2 * 512 * 2) / 16; i += 512) {
            unsigned long long g = (unsigned long long)(src + (size_t)i * 16);
            unsigned d = ldsBase + i * 16;
            asm volatile("global_load_async_to_lds_b128 %0, %1, off"
                         :: "v"(d), "v"(g) : "memory");
        }
        for (int i = tid; i < BSZ * H2; i += 512) { hS[i] = 0; cS[i] = 0.f; }
        asm volatile("s_wait_asynccnt 0x0" ::: "memory");
    }
    __syncthreads();

    const int hiA  = (lane >> 4) << 3;
    const int colC = lane & 15;

    for (int step = 0; step < LSEQ; ++step) {
        const int l = dir ? (LSEQ - 1 - step) : step;

        // prefetch next step's gx slice (contiguous 128KB) into cache
        if (step + 1 < LSEQ) {
            const int ln = dir ? (LSEQ - 2 - step) : (step + 1);
            const char* pf = (const char*)(gx + (((size_t)ln * 2 + dir) * 128) * 256)
                             + (size_t)tid * 256;
            __builtin_prefetch(pf, 0, 1);
            __builtin_prefetch(pf + 128, 0, 1);
        }

        // ---- g = gx[l] + h @ Whh^T : M=64, N=512, K=128 (128 WMMA tiles) ----
#pragma unroll
        for (int t8 = 0; t8 < 8; ++t8) {
            const int tile = wave * 8 + t8;
            const int mT = tile & 3, nT = tile >> 2;
            const int n0 = nT * 16;
            // accumulator init: fragment-tiled gx, 2x b128
            const size_t tileIdx = (((size_t)l * 2 + dir) * 32 + nT) * 4 + mT;
            const float* gp = gx + tileIdx * 256 + lane * 8;
            Frag32 cf;
            cf.u[0] = *(const u32x4*)(gp);
            cf.u[1] = *(const u32x4*)(gp + 4);
            v8f c = cf.f;
#pragma unroll
            for (int kk = 0; kk < 4; ++kk) {
                Frag16 a, bb;
                const unsigned short* ap = hS + (mT * 16 + colC) * H2 + kk * 32 + hiA;
                a.u[0] = *(const u32x4*)(ap);        // ds_load_b128
                a.u[1] = *(const u32x4*)(ap + 16);
                const unsigned short* bp = wS + (n0 + colC) * H2 + kk * 32 + hiA;
                bb.u[0] = *(const u32x4*)(bp);
                bb.u[1] = *(const u32x4*)(bp + 16);
                c = __builtin_amdgcn_wmma_f32_16x16x32_bf16(
                    false, a.v, false, bb.v, (short)0, c, false, false);
            }
            const int rowOfs = (lane >> 4) << 3;
#pragma unroll
            for (int r = 0; r < 8; ++r)
                gS[(mT * 16 + r + rowOfs) * 512 + n0 + colC] = c[r];
        }
        __syncthreads();

        // ---- gate nonlinearity + state update (8192 elems / 512 thr) ----
#pragma unroll 4
        for (int it = 0; it < 16; ++it) {
            const int idx = tid * 16 + it;
            const int b = idx >> 7, k = idx & 127;
            const float gi = gS[b * 512 + k];
            const float gf = gS[b * 512 + 128 + k];
            const float gg = gS[b * 512 + 256 + k];
            const float go = gS[b * 512 + 384 + k];
            const float i_ = 1.f / (1.f + expf(-gi));
            const float f_ = 1.f / (1.f + expf(-gf));
            const float g_ = tanhf(gg);
            const float o_ = 1.f / (1.f + expf(-go));
            const float cOld = cS[b * H2 + k];
            const float hOld = bf16_to_f32(hS[b * H2 + k]);
            const float cNew = f_ * cOld + i_ * g_;
            const float hNew = o_ * tanhf(cNew);
            const int   mt   = pm[b * LSEQ + l];
            cS[b * H2 + k] = mt ? cNew : cOld;
            hS[b * H2 + k] = f32_to_bf16(mt ? hNew : hOld);
            lstm_out[((size_t)l * BSZ + b) * 256 + dir * H2 + k] = mt ? hNew : 0.f;
        }
        __syncthreads();
    }
}

// ---------------------------------------------------------------------------
// Kernel 5: emissions[m][t] = lstm_out[m][:] . W_out[t][:] + b_out[t]
// One thread per row m computes all 9 tags; W_out cached in LDS.
// ---------------------------------------------------------------------------
__launch_bounds__(256)
__global__ void emis_kernel(const float* __restrict__ lo,
                            const float* __restrict__ W_out,
                            const float* __restrict__ b_out,
                            float* __restrict__ emi) {
    __shared__ float wS[NT * 256];
    for (int i = threadIdx.x; i < NT * 256; i += 256) wS[i] = W_out[i];
    __syncthreads();
    const int m = blockIdx.x * 256 + threadIdx.x;
    float acc[NT];
#pragma unroll
    for (int t = 0; t < NT; ++t) acc[t] = b_out[t];
    const float* x = lo + (size_t)m * 256;
#pragma unroll 4
    for (int e = 0; e < 256; ++e) {
        const float xe = x[e];
#pragma unroll
        for (int t = 0; t < NT; ++t) acc[t] += xe * wS[t * 256 + e];
    }
#pragma unroll
    for (int t = 0; t < NT; ++t) emi[(size_t)m * NT + t] = acc[t];
}

// ---------------------------------------------------------------------------
// Kernel 6: CRF numerator + forward logZ + loss + Viterbi decode.
// ---------------------------------------------------------------------------
__launch_bounds__(64)
__global__ void crf_kernel(const int* __restrict__ tags,
                           const int* __restrict__ pm,
                           const float* __restrict__ emi,
                           const float* __restrict__ startT,
                           const float* __restrict__ endT,
                           const float* __restrict__ trans,
                           unsigned char* __restrict__ hist,
                           float* __restrict__ out) {
    __shared__ float sT[NT * NT], sS[NT], sE[NT], red[BSZ];
    const int tid = threadIdx.x;
    for (int i = tid; i < NT * NT; i += BSZ) sT[i] = trans[i];
    if (tid < NT) { sS[tid] = startT[tid]; sE[tid] = endT[tid]; }
    __syncthreads();
    const int b = tid;

    int len = 0;
    for (int l = 0; l < LSEQ; ++l) len += pm[b * LSEQ + l];

    // numerator
    int prev = tags[b * LSEQ];
    float num = sS[prev] + emi[(size_t)b * NT + prev];
    for (int l = 1; l < LSEQ; ++l) {
        int tg = tags[b * LSEQ + l];
        if (pm[b * LSEQ + l])
            num += sT[prev * NT + tg] + emi[((size_t)l * BSZ + b) * NT + tg];
        prev = tg;
    }
    num += sE[tags[b * LSEQ + (len - 1)]];

    // forward algorithm
    float sc[NT];
#pragma unroll
    for (int j = 0; j < NT; ++j) sc[j] = sS[j] + emi[(size_t)b * NT + j];
    for (int l = 1; l < LSEQ; ++l) {
        if (!pm[b * LSEQ + l]) continue;
        const float* el = emi + ((size_t)l * BSZ + b) * NT;
        float ns[NT];
#pragma unroll
        for (int j = 0; j < NT; ++j) {
            float mx = -1e30f;
#pragma unroll
            for (int i = 0; i < NT; ++i) mx = fmaxf(mx, sc[i] + sT[i * NT + j]);
            float s = 0.f;
#pragma unroll
            for (int i = 0; i < NT; ++i) s += expf(sc[i] + sT[i * NT + j] - mx);
            ns[j] = mx + logf(s) + el[j];
        }
#pragma unroll
        for (int j = 0; j < NT; ++j) sc[j] = ns[j];
    }
    float mx = -1e30f;
#pragma unroll
    for (int j = 0; j < NT; ++j) mx = fmaxf(mx, sc[j] + sE[j]);
    float s = 0.f;
#pragma unroll
    for (int j = 0; j < NT; ++j) s += expf(sc[j] + sE[j] - mx);
    const float logZ = mx + logf(s);

    red[tid] = num - logZ;
    __syncthreads();
    if (tid == 0) {
        float acc = 0.f;
        for (int i = 0; i < BSZ; ++i) acc += red[i];
        out[0] = -acc / (float)BSZ;
    }

    // Viterbi
    float vs[NT];
#pragma unroll
    for (int j = 0; j < NT; ++j) vs[j] = sS[j] + emi[(size_t)b * NT + j];
    unsigned char* hb = hist + (size_t)b * (LSEQ - 1) * NT;
    for (int l = 1; l < LSEQ; ++l) {
        const float* el = emi + ((size_t)l * BSZ + b) * NT;
        float nv[NT];
#pragma unroll
        for (int j = 0; j < NT; ++j) {
            float best = -1e30f; int bi = 0;
#pragma unroll
            for (int i = 0; i < NT; ++i) {
                float v = vs[i] + sT[i * NT + j];
                if (v > best) { best = v; bi = i; }
            }
            nv[j] = best + el[j];
            hb[(l - 1) * NT + j] = (unsigned char)bi;
        }
#pragma unroll
        for (int j = 0; j < NT; ++j) vs[j] = nv[j];
    }
    float best = -1e30f; int tag = 0;
#pragma unroll
    for (int j = 0; j < NT; ++j) {
        float v = vs[j] + sE[j];
        if (v > best) { best = v; tag = j; }
    }
    for (int l = LSEQ - 1; l >= 1; --l) {
        out[1 + b * LSEQ + l] = (float)(tag * pm[b * LSEQ + l]);
        tag = hb[(l - 1) * NT + tag];
    }
    out[1 + b * LSEQ] = (float)(tag * pm[b * LSEQ]);
}

// ---------------------------------------------------------------------------
extern "C" void kernel_launch(void* const* d_in, const int* in_sizes, int n_in,
                              void* d_out, int out_size, void* d_ws, size_t ws_size,
                              hipStream_t stream) {
    (void)in_sizes; (void)n_in; (void)out_size; (void)ws_size;
    const int*   sentence = (const int*)  d_in[0];
    const int*   tags     = (const int*)  d_in[1];
    const int*   pmask    = (const int*)  d_in[2];
    // d_in[3] = batch_sequence_max_len (== 512, compile-time here)
    const float* emb      = (const float*)d_in[4];
    const float* w_ih_f   = (const float*)d_in[5];
    const float* w_hh_f   = (const float*)d_in[6];
    const float* b_f      = (const float*)d_in[7];
    const float* w_ih_b   = (const float*)d_in[8];
    const float* w_hh_b   = (const float*)d_in[9];
    const float* b_b      = (const float*)d_in[10];
    const float* W_out    = (const float*)d_in[11];
    const float* b_out    = (const float*)d_in[12];
    const float* startT   = (const float*)d_in[13];
    const float* endT     = (const float*)d_in[14];
    const float* trans    = (const float*)d_in[15];
    float* out = (float*)d_out;

    char* ws = (char*)d_ws;
    unsigned short* X     = (unsigned short*)(ws + OFF_X);
    unsigned short* WihT  = (unsigned short*)(ws + OFF_WIH);
    unsigned short* WhhT  = (unsigned short*)(ws + OFF_WHH);
    float*          biasG = (float*)         (ws + OFF_BIAS);
    float*          gx    = (float*)         (ws + OFF_GX);
    float*          lo    = (float*)         (ws + OFF_LO);
    float*          emi   = (float*)         (ws + OFF_EMI);
    unsigned char*  hist  = (unsigned char*) (ws + OFF_HIST);

    // 1) weight prep (transpose + bf16)
    {
        int total = EMB * NG + 2 * H2 * 512 + NG;
        prep_kernel<<<(total + 255) / 256, 256, 0, stream>>>(
            w_ih_f, w_hh_f, b_f, w_ih_b, w_hh_b, b_b, WihT, WhhT, biasG);
    }
    // 2) embedding gather -> bf16
    embed_kernel<<<ML, EMB, 0, stream>>>(sentence, emb, X);
    // 3) big input GEMM (WMMA bf16, f32 accum), fragment-tiled output
    gemm_gx_kernel<<<dim3(ML / 32, 2), 256, 0, stream>>>(X, WihT, biasG, gx);
    // 4) sequential bidirectional LSTM (1 WGP per direction, LDS-resident)
    {
        size_t lds = 16384 + 32768 + 131072 + 131072;   // 304 KB of 320 KB
        lstm_kernel<<<dim3(2), 512, lds, stream>>>(gx, WhhT, pmask, lo);
    }
    // 5) emissions
    emis_kernel<<<ML / 256, 256, 0, stream>>>(lo, W_out, b_out, emi);
    // 6) CRF loss + Viterbi decode
    crf_kernel<<<1, BSZ, 0, stream>>>(tags, pmask, emi, startT, endT, trans, hist, out);
}
